// GraphConvolution_60069412601881
// MI455X (gfx1250) — matrix operand order repaired and checked by reading, not compiled, not run on hardware
//
#include <hip/hip_runtime.h>
#include <math.h>

#define D 64
#define MIN_NORM   1e-15f
#define PROJ_EPS   1e-5f
#define ATANH_CLIP (1.0f - 1e-7f)

typedef __bf16 bf16x16 __attribute__((ext_vector_type(16)));
typedef float  f32x8   __attribute__((ext_vector_type(8)));

__device__ __forceinline__ float artanh_c(float v) {
    v = fminf(fmaxf(v, -ATANH_CLIP), ATANH_CLIP);
    return atanhf(v);
}

__device__ __forceinline__ float wave_sum(float v) {
    // wave32 butterfly reduction
    v += __shfl_xor(v, 16, 32);
    v += __shfl_xor(v,  8, 32);
    v += __shfl_xor(v,  4, 32);
    v += __shfl_xor(v,  2, 32);
    v += __shfl_xor(v,  1, 32);
    return v;
}

// ---------------------------------------------------------------------------
// Kernel 1: mx = x @ W^T via V_WMMA_F32_16X16X32_BF16 with hi/lo bf16 split.
// One wave32 computes a 16-row stripe (16x64 output), 4 col tiles, 24 WMMAs.
// ---------------------------------------------------------------------------
__global__ void __launch_bounds__(256)
gemm_wmma_bf16x3(const float* __restrict__ x, const float* __restrict__ W,
                 float* __restrict__ mx, int nrows) {
    const int lane = threadIdx.x & 31;
    const int wave = blockIdx.x * (blockDim.x >> 5) + (threadIdx.x >> 5);
    const int i0   = wave * 16;
    if (i0 >= nrows) return;
    const int half = lane >> 4;     // 0 or 1
    const int m    = lane & 15;

    // ---- A fragments: 16x32 bf16 tiles, hi/lo split, for K-chunks 0..31,32..63
    // ISA layout: lane(half=0) holds K={0..7,16..23}; half=1 holds K={8..15,24..31}
    int arow = i0 + m; if (arow >= nrows) arow = nrows - 1;
    const float* xrow = x + (size_t)arow * D;
    bf16x16 a_hi[2], a_lo[2];
#pragma unroll
    for (int kc = 0; kc < 2; ++kc) {
#pragma unroll
        for (int e = 0; e < 16; ++e) {
            int K = 32 * kc + (e & 7) + ((e >> 3) << 4) + (half << 3);
            float v = xrow[K];
            __bf16 h = (__bf16)v;
            a_hi[kc][e] = h;
            a_lo[kc][e] = (__bf16)(v - (float)h);
        }
    }

#pragma unroll
    for (int jt = 0; jt < 4; ++jt) {
        const int j0 = jt * 16;
        f32x8 c = {0.f, 0.f, 0.f, 0.f, 0.f, 0.f, 0.f, 0.f};
#pragma unroll
        for (int kc = 0; kc < 2; ++kc) {
            // B = W^T tile (32x16): B[k, n] = W[j0+n, k].
            // ISA layout: lanes 0-15 hold K={0..7,16..23}, lanes 16-31 K={8..15,24..31}; col = m.
            bf16x16 b_hi, b_lo;
            const float* wrow = W + (size_t)(j0 + m) * D;
#pragma unroll
            for (int e = 0; e < 16; ++e) {
                int K = 32 * kc + (e & 7) + ((e >> 3) << 4) + (half << 3);
                float v = wrow[K];
                __bf16 h = (__bf16)v;
                b_hi[e] = h;
                b_lo[e] = (__bf16)(v - (float)h);
            }
            c = __builtin_amdgcn_wmma_f32_16x16x32_bf16(false, a_hi[kc], false, b_hi,
                                                        (short)0, c, false, false);
            c = __builtin_amdgcn_wmma_f32_16x16x32_bf16(false, a_lo[kc], false, b_hi,
                                                        (short)0, c, false, false);
            c = __builtin_amdgcn_wmma_f32_16x16x32_bf16(false, a_hi[kc], false, b_lo,
                                                        (short)0, c, false, false);
        }
        // D layout: VGPR r -> row r + 8*half, col = m
#pragma unroll
        for (int r = 0; r < 8; ++r) {
            int row = i0 + r + 8 * half;
            if (row < nrows) mx[(size_t)row * D + j0 + m] = c[r];
        }
    }
}

// ---------------------------------------------------------------------------
// Kernel 2: per-node mobius_matvec tail + logmap0  -> hidden
// One wave32 per node, 2 elements per lane.
// ---------------------------------------------------------------------------
__global__ void __launch_bounds__(256)
node_log0(const float* __restrict__ x, const float* __restrict__ mx,
          float* __restrict__ hidden, int n) {
    const int lane = threadIdx.x & 31;
    const int node = blockIdx.x * (blockDim.x >> 5) + (threadIdx.x >> 5);
    if (node >= n) return;
    const float* xi = x  + (size_t)node * D;
    const float* mi = mx + (size_t)node * D;
    float x0 = xi[lane], x1 = xi[lane + 32];
    float m0 = mi[lane], m1 = mi[lane + 32];
    float xs = wave_sum(x0 * x0 + x1 * x1);
    float ms = wave_sum(m0 * m0 + m1 * m1);
    float x_n  = fmaxf(sqrtf(xs), MIN_NORM);
    float mx_n = fmaxf(sqrtf(ms), MIN_NORM);
    float t = tanhf(mx_n / x_n * artanh_c(x_n));
    float s = (ms == 0.0f) ? 0.0f : (t / mx_n);   // all(mx==0) -> 0
    float r0 = s * m0, r1 = s * m1;
    float ps  = wave_sum(r0 * r0 + r1 * r1);
    float p_n = fmaxf(sqrtf(ps), MIN_NORM);
    float hs  = artanh_c(p_n) / p_n;
    float* hi = hidden + (size_t)node * D;
    hi[lane]      = hs * r0;
    hi[lane + 32] = hs * r1;
}

// ---------------------------------------------------------------------------
// Kernel 3a: zero support accumulator
// ---------------------------------------------------------------------------
__global__ void __launch_bounds__(256)
zero_f32(float* __restrict__ p, size_t n) {
    size_t i = (size_t)blockIdx.x * blockDim.x + threadIdx.x;
    if (i < n) p[i] = 0.0f;
}

// ---------------------------------------------------------------------------
// Kernel 3b: SpMM scatter: support[rows[e]] += vals[e] * hidden[cols[e]]
// 8 threads per edge, float4 gathers, global_atomic_add_f32 scatter.
// ---------------------------------------------------------------------------
__global__ void __launch_bounds__(256)
spmm_scatter(const float* __restrict__ hidden, const float* __restrict__ vals,
             const int* __restrict__ rows, const int* __restrict__ cols,
             float* __restrict__ support, int E) {
    long long gid = (long long)blockIdx.x * blockDim.x + threadIdx.x;
    int e = (int)(gid >> 3);
    if (e >= E) return;
    int part = (int)(gid & 7);
    int r = rows[e], c = cols[e];
    float v = vals[e];
    const float4* hsrc = (const float4*)(hidden + (size_t)c * D + part * 8);
    float4 h0 = hsrc[0], h1 = hsrc[1];
    float* dst = support + (size_t)r * D + part * 8;
    atomicAdd(dst + 0, v * h0.x); atomicAdd(dst + 1, v * h0.y);
    atomicAdd(dst + 2, v * h0.z); atomicAdd(dst + 3, v * h0.w);
    atomicAdd(dst + 4, v * h1.x); atomicAdd(dst + 5, v * h1.y);
    atomicAdd(dst + 6, v * h1.z); atomicAdd(dst + 7, v * h1.w);
}

// ---------------------------------------------------------------------------
// Kernel 4: expmap0 -> relu(logmap0) -> expmap0 -> proj   (sc = 1)
// ---------------------------------------------------------------------------
__global__ void __launch_bounds__(256)
node_exp_act(const float* __restrict__ support, float* __restrict__ out, int n) {
    const int lane = threadIdx.x & 31;
    const int node = blockIdx.x * (blockDim.x >> 5) + (threadIdx.x >> 5);
    if (node >= n) return;
    const float* si = support + (size_t)node * D;
    float s0 = si[lane], s1 = si[lane + 32];
    // expmap0
    float us  = wave_sum(s0 * s0 + s1 * s1);
    float u_n = fmaxf(sqrtf(us), MIN_NORM);
    float psc = tanhf(u_n) / u_n;
    float p0 = psc * s0, p1 = psc * s1;
    // relu(logmap0)
    float p2s = wave_sum(p0 * p0 + p1 * p1);
    float pn2 = fmaxf(sqrtf(p2s), MIN_NORM);
    float lsc = artanh_c(pn2) / pn2;
    float xt0 = fmaxf(lsc * p0, 0.0f), xt1 = fmaxf(lsc * p1, 0.0f);
    // expmap0
    float xts  = wave_sum(xt0 * xt0 + xt1 * xt1);
    float xt_n = fmaxf(sqrtf(xts), MIN_NORM);
    float esc  = tanhf(xt_n) / xt_n;
    float o0 = esc * xt0, o1 = esc * xt1;
    // proj clip
    float os  = wave_sum(o0 * o0 + o1 * o1);
    float o_n = fmaxf(sqrtf(os), MIN_NORM);
    const float max_n = 1.0f - PROJ_EPS;
    if (o_n > max_n) { float f = max_n / o_n; o0 *= f; o1 *= f; }
    float* oi = out + (size_t)node * D;
    oi[lane]      = o0;
    oi[lane + 32] = o1;
}

// ---------------------------------------------------------------------------
extern "C" void kernel_launch(void* const* d_in, const int* in_sizes, int n_in,
                              void* d_out, int out_size, void* d_ws, size_t ws_size,
                              hipStream_t stream) {
    const float* x    = (const float*)d_in[0];
    const float* W    = (const float*)d_in[1];
    const float* vals = (const float*)d_in[2];
    const int*   rows = (const int*)d_in[3];
    const int*   cols = (const int*)d_in[4];
    const int N = in_sizes[0] / D;
    const int E = in_sizes[2];

    float* mx      = (float*)d_ws;                 // N*D
    float* hidden  = mx     + (size_t)N * D;       // N*D
    float* support = hidden + (size_t)N * D;       // N*D

    {   // GEMM: one wave per 16 rows, 8 waves per block
        int tiles  = (N + 15) / 16;
        int blocks = (tiles + 7) / 8;
        gemm_wmma_bf16x3<<<blocks, 256, 0, stream>>>(x, W, mx, N);
    }
    {   // per-node poincare-linear tail + logmap0
        int blocks = (N + 7) / 8;
        node_log0<<<blocks, 256, 0, stream>>>(x, mx, hidden, N);
    }
    {   // zero accumulator
        size_t tot = (size_t)N * D;
        int blocks = (int)((tot + 255) / 256);
        zero_f32<<<blocks, 256, 0, stream>>>(support, tot);
    }
    {   // sparse aggregation
        long long threads = (long long)E * 8;
        int blocks = (int)((threads + 255) / 256);
        spmm_scatter<<<blocks, 256, 0, stream>>>(hidden, vals, rows, cols, support, E);
    }
    {   // expmap0 -> act -> proj
        int blocks = (N + 7) / 8;
        node_exp_act<<<blocks, 256, 0, stream>>>(support, (float*)d_out, N);
    }
}
